// LinearFP8_50122268345047
// MI455X (gfx1250) — 8-bit matrix operand order repaired and checked by reading, compiled, not run on hardware
//
#include <hip/hip_runtime.h>
#include <hip/hip_bf16.h>


typedef __attribute__((ext_vector_type(16))) int   v16i;
typedef __attribute__((ext_vector_type(8)))  float v8f;

#define FP8_MAX 448.0f

__device__ __forceinline__ unsigned char fp8_e4m3_sw(float x) {
  unsigned u  = __float_as_uint(x);
  unsigned s  = (u >> 24) & 0x80u;
  unsigned au = u & 0x7fffffffu;
  if (au >= 0x43e00000u) return (unsigned char)(s | 0x7e);
  if (au <  0x3a000000u) return (unsigned char)s;
  unsigned lsb = (au >> 20) & 1u;
  au += 0x0007ffffu + lsb;
  int exp = (int)(au >> 23) - 127;
  unsigned man = (au >> 20) & 7u;
  if (exp < -6) {
    int sh = -6 - exp;
    unsigned d = (man | 8u) >> (sh > 4 ? 4 : sh);
    return (unsigned char)(s | d);
  }
  return (unsigned char)(s | ((unsigned)(exp + 7) << 3) | man);
}

__device__ __forceinline__ unsigned pack4_fp8(float a, float b, float c, float d) {
#if defined(__has_builtin) && __has_builtin(__builtin_amdgcn_cvt_pk_fp8_f32)
  int lo = __builtin_amdgcn_cvt_pk_fp8_f32(a, b, 0, false);
  int hi = __builtin_amdgcn_cvt_pk_fp8_f32(c, d, lo, true);
  return (unsigned)hi;
#else
  return (unsigned)fp8_e4m3_sw(a) | ((unsigned)fp8_e4m3_sw(b) << 8) |
         ((unsigned)fp8_e4m3_sw(c) << 16) | ((unsigned)fp8_e4m3_sw(d) << 24);
#endif
}

__device__ __forceinline__ float bf16_bits_to_f(unsigned hi16) {
  return __uint_as_float(hi16 << 16);
}

__global__ void __launch_bounds__(256)
quantize_rows_kernel(const __hip_bfloat16* __restrict__ x,
                     unsigned char* __restrict__ xq,
                     float* __restrict__ xscale, int K) {
  const int row  = blockIdx.x;
  const int tid  = threadIdx.x;
  const int lane = tid & 31;
  const int wave = tid >> 5;
  const size_t rbase = (size_t)row * (size_t)K;
  const unsigned short* xr = (const unsigned short*)(x + rbase);

  float amax = 0.0f;
  for (int base = tid * 16; base < K; base += 256 * 16) {
    uint4 u0 = *(const uint4*)(xr + base);
    uint4 u1 = *(const uint4*)(xr + base + 8);
    unsigned ws[8] = {u0.x, u0.y, u0.z, u0.w, u1.x, u1.y, u1.z, u1.w};
#pragma unroll
    for (int q = 0; q < 8; ++q) {
      amax = fmaxf(amax, fabsf(bf16_bits_to_f(ws[q] & 0xffffu)));
      amax = fmaxf(amax, fabsf(bf16_bits_to_f(ws[q] >> 16)));
    }
  }
#pragma unroll
  for (int off = 16; off; off >>= 1)
    amax = fmaxf(amax, __shfl_xor(amax, off, 32));

  __shared__ float sred[8];
  if (lane == 0) sred[wave] = amax;
  __syncthreads();
  float rmax = fmaxf(fmaxf(fmaxf(sred[0], sred[1]), fmaxf(sred[2], sred[3])),
                     fmaxf(fmaxf(sred[4], sred[5]), fmaxf(sred[6], sred[7])));
  rmax = fmaxf(rmax, __bfloat162float(__float2bfloat16(1e-4f)));
  const float inv_scale = FP8_MAX / rmax;

  if (tid == 0) xscale[row] = rmax * (1.0f / FP8_MAX);

  for (int base = tid * 16; base < K; base += 256 * 16) {
    uint4 u0 = *(const uint4*)(xr + base);
    uint4 u1 = *(const uint4*)(xr + base + 8);
    unsigned ws[8] = {u0.x, u0.y, u0.z, u0.w, u1.x, u1.y, u1.z, u1.w};
    float f[16];
#pragma unroll
    for (int q = 0; q < 8; ++q) {
      f[2 * q]     = bf16_bits_to_f(ws[q] & 0xffffu) * inv_scale;
      f[2 * q + 1] = bf16_bits_to_f(ws[q] >> 16) * inv_scale;
    }
    uint4 o;
    o.x = pack4_fp8(f[0],  f[1],  f[2],  f[3]);
    o.y = pack4_fp8(f[4],  f[5],  f[6],  f[7]);
    o.z = pack4_fp8(f[8],  f[9],  f[10], f[11]);
    o.w = pack4_fp8(f[12], f[13], f[14], f[15]);
    *(uint4*)(xq + rbase + base) = o;
  }
}

__global__ void __launch_bounds__(256)
gemm_fp8_wmma_kernel(const unsigned char* __restrict__ xq,
                     const float* __restrict__ xs,
                     const unsigned char* __restrict__ w,
                     const float* __restrict__ sw,
                     const __hip_bfloat16* __restrict__ bias,
                     __hip_bfloat16* __restrict__ out,
                     int M, int N, int K) {
  const int lane = threadIdx.x & 31;
  const int wave = threadIdx.x >> 5;
  const int wm   = wave >> 2;
  const int wn   = wave & 3;
  const int m0   = blockIdx.y * 128 + wm * 64;
  const int n0   = blockIdx.x * 256 + wn * 64;
  const int lmod = lane & 15;
  const int lhi  = lane >> 4;

  v8f acc[4][4];
#pragma unroll
  for (int i = 0; i < 4; ++i)
#pragma unroll
    for (int j = 0; j < 4; ++j) {
      v8f z = {0.f, 0.f, 0.f, 0.f, 0.f, 0.f, 0.f, 0.f};
      acc[i][j] = z;
    }

  for (int k0 = 0; k0 < K; k0 += 128) {
    v16i A[4];
#pragma unroll
    for (int i = 0; i < 4; ++i) {
      const unsigned char* ap =
          xq + (size_t)(m0 + i * 16 + lmod) * (size_t)K + (size_t)(k0 + lhi * 8);
#pragma unroll
      for (int j = 0; j < 8; ++j) {
        unsigned long long d = *(const unsigned long long*)(ap + j * 16);
        A[i][2 * j]     = (int)(unsigned)(d & 0xffffffffu);
        A[i][2 * j + 1] = (int)(unsigned)(d >> 32);
      }
    }
    v16i Bm[4];
#pragma unroll
    for (int j = 0; j < 4; ++j) {
      const unsigned char* bp =
          w + (size_t)(n0 + j * 16 + lmod) * (size_t)K + (size_t)(k0 + lhi * 8);
#pragma unroll
      for (int g = 0; g < 8; ++g) {
        unsigned long long t = *(const unsigned long long*)(bp + g * 16);
        Bm[j][2 * g]     = (int)(unsigned)(t & 0xffffffffu);
        Bm[j][2 * g + 1] = (int)(unsigned)(t >> 32);
      }
    }
    if (k0 + 128 < K) {
      __builtin_prefetch(xq + (size_t)(m0 + lmod) * (size_t)K + k0 + 128, 0, 3);
      __builtin_prefetch(w + (size_t)(n0 + lmod) * (size_t)K + k0 + 128, 0, 3);
    }
#pragma unroll
    for (int i = 0; i < 4; ++i)
#pragma unroll
      for (int j = 0; j < 4; ++j)
        acc[i][j] = __builtin_amdgcn_wmma_f32_16x16x128_fp8_fp8(
            A[i], Bm[j], (short)0, acc[i][j], false, false);
  }

  float swv[4], bv[4];
#pragma unroll
  for (int j = 0; j < 4; ++j) {
    int n  = n0 + j * 16 + lmod;
    swv[j] = sw[n];
    bv[j]  = __bfloat162float(bias[n]);
  }
#pragma unroll
  for (int i = 0; i < 4; ++i) {
#pragma unroll
    for (int r = 0; r < 8; ++r) {
      const int m = m0 + i * 16 + r + lhi * 8;
      const float xsv = xs[m];
#pragma unroll
      for (int j = 0; j < 4; ++j) {
        const int n = n0 + j * 16 + lmod;
        float v = acc[i][j][r] * xsv * swv[j];
        float o = __bfloat162float(__float2bfloat16(v)) + bv[j];
        out[(size_t)m * (size_t)N + (size_t)n] = __float2bfloat16(o);
      }
    }
  }
}

extern "C" void kernel_launch(void* const* d_in, const int* in_sizes, int n_in,
                              void* d_out, int out_size, void* d_ws, size_t ws_size,
                              hipStream_t stream) {
  (void)n_in; (void)out_size; (void)ws_size;
  const __hip_bfloat16* x    = (const __hip_bfloat16*)d_in[0];
  const unsigned char*  wq   = (const unsigned char*)d_in[1];
  const float*          sw   = (const float*)d_in[2];
  const __hip_bfloat16* bias = (const __hip_bfloat16*)d_in[3];
  __hip_bfloat16*       out  = (__hip_bfloat16*)d_out;

  const int N = in_sizes[2];
  const int K = in_sizes[1] / N;
  const int M = in_sizes[0] / K;

  unsigned char* xq = (unsigned char*)d_ws;
  float* xs = (float*)((char*)d_ws + (size_t)M * (size_t)K);

  quantize_rows_kernel<<<dim3(M), dim3(256), 0, stream>>>(x, xq, xs, K);

  dim3 grid(N / 256, M / 128);
  gemm_fp8_wmma_kernel<<<grid, dim3(256), 0, stream>>>(xq, xs, wq, sw, bias, out,
                                                       M, N, K);
}
